// MoEnhanceTaskBlock_83777632076364
// MI455X (gfx1250) — matrix operand order repaired and checked by reading, not compiled, not run on hardware
//
#include <hip/hip_runtime.h>
#include <math.h>

typedef _Float16 f16;
typedef __attribute__((ext_vector_type(16))) _Float16 v16h;
typedef __attribute__((ext_vector_type(8)))  float    v8f;
typedef unsigned int u32;

#define B_   4
#define N_   1024
#define D_   768
#define H_   12
#define HD_  64
#define EA_  24
#define EF_  8
#define HS_  1536
#define TOK  (B_*N_)   // 4096

// ---------------------------------------------------------------- helpers

static __device__ __forceinline__ float gelu_f(float x){
  return 0.5f*x*(1.0f + erff(x*0.70710678118654752f));
}

static __device__ __forceinline__ v8f wmma16(v16h a, v16h b, v8f c){
  // D(16x16,f32) = A(16x32,f16) x B(32x16,f16) + C
  return __builtin_amdgcn_wmma_f32_16x16x32_f16(false, a, false, b, (short)0, c, false, false);
}

// A fragment: lane L holds row M=L&15; element e -> K = 16*(e>>3) + 8*(L>>4) + (e&7)
// (two contiguous 8-half chunks per lane -> vectorizes to 2x b128 loads)
static __device__ __forceinline__ v16h frag_a_f16(const f16* base, int ld){
  int lane = threadIdx.x & 31; int row = lane & 15; int khi = (lane >> 4) << 3;
  v16h a;
#pragma unroll
  for (int e = 0; e < 16; ++e){
    int k = ((e >> 3) << 4) + khi + (e & 7);
    a[e] = base[(size_t)row * ld + k];
  }
  return a;
}

// B fragment from K-contiguous transposed LDS tile laid out [N][32]:
// lane L holds col N=L&15, K = 16*(e>>3) + 8*(L>>4) + (e&7)  -> 2x 8-half chunks = 2x ds_load_b128
static __device__ __forceinline__ v16h frag_b_ldsT(const f16* baseT){
  int lane = threadIdx.x & 31; int c = lane & 15; int k0 = (lane >> 4) << 3;
  const f16* p = baseT + (size_t)c * 32 + k0;
  v16h b;
#pragma unroll
  for (int e = 0; e < 16; ++e) b[e] = p[((e >> 3) << 4) + (e & 7)];
  return b;
}

// B fragment (row-major N x K f16 in GLOBAL memory, ld stride): K = 16*(e>>3) + 8*(L>>4) + (e&7) per lane
// only used where memory is K-contiguous already (k^T reads in attention)
static __device__ __forceinline__ v16h frag_bt_f16(const f16* base, int ld){
  int lane = threadIdx.x & 31; int c = lane & 15; int k0 = (lane >> 4) << 3;
  v16h b;
#pragma unroll
  for (int e = 0; e < 16; ++e) b[e] = base[(size_t)c * ld + k0 + ((e >> 3) << 4) + (e & 7)];
  return b;
}

// Per-wave staging of a 32(K) x 16(N) f32 weight tile:
// coalesced row reads (64B/lane) -> LDS transposed [16][32] f16 -> vector fragment reads.
// Same-wave LDS store->load ordering closed with explicit s_wait_dscnt.
static __device__ __forceinline__ v16h frag_b_f32_stage(const float* g, int ld, f16* wb){
  int lane = threadIdx.x & 31;
  const float* r = g + (size_t)lane * ld;   // row kk+lane, 16 contiguous cols
#pragma unroll
  for (int n = 0; n < 16; ++n) wb[n * 32 + lane] = (f16)r[n];
  __builtin_prefetch(r + (size_t)32 * ld, 0, 1);   // next K-step, same lane row
  asm volatile("s_wait_dscnt 0x0" ::: "memory");
  return frag_b_ldsT(wb);
}

static __device__ __forceinline__ float block_sum256(float v, float* red){
  int t = threadIdx.x;
  red[t] = v; __syncthreads();
  for (int o = 128; o > 0; o >>= 1){ if (t < o) red[t] += red[t + o]; __syncthreads(); }
  float r = red[0]; __syncthreads();
  return r;
}

// ---------------------------------------------------------------- LN + routing (attention, top-12 of 24)

__global__ __launch_bounds__(256) void ln_route1_kernel(
    const float* __restrict__ x, const float* __restrict__ g, const float* __restrict__ bb,
    const float* __restrict__ gateA, const int* __restrict__ taskp,
    f16* __restrict__ h1, int* __restrict__ topi, float* __restrict__ topv)
{
  __shared__ float hrow[D_];
  __shared__ float red[256];
  __shared__ float logits[EA_];
  int tok = blockIdx.x;
  const float* xr = x + (size_t)tok * D_;

  float s = 0.f;
  for (int i = threadIdx.x; i < D_; i += 256) s += xr[i];
  float mean = block_sum256(s, red) * (1.0f / D_);
  float vs = 0.f;
  for (int i = threadIdx.x; i < D_; i += 256){ float d = xr[i] - mean; vs += d * d; }
  float var = block_sum256(vs, red) * (1.0f / D_);
  float inv = rsqrtf(var + 1e-5f);
  for (int i = threadIdx.x; i < D_; i += 256){
    float h = (xr[i] - mean) * inv * g[i] + bb[i];
    hrow[i] = h;
    h1[(size_t)tok * D_ + i] = (f16)h;
  }
  __syncthreads();

  int task = taskp[0];
  const float* gA = gateA + (size_t)task * D_ * EA_;
  if (threadIdx.x < EA_){
    float acc = 0.f;
    for (int d = 0; d < D_; ++d) acc += hrow[d] * gA[(size_t)d * EA_ + threadIdx.x];
    logits[threadIdx.x] = acc;
  }
  __syncthreads();
  if (threadIdx.x == 0){
    float lg[EA_]; int sel[H_]; float sv[H_];
    for (int i = 0; i < EA_; ++i) lg[i] = logits[i];
    for (int k = 0; k < H_; ++k){
      int bi = 0; float bv = -1e30f;
      for (int i = 0; i < EA_; ++i) if (lg[i] > bv){ bv = lg[i]; bi = i; }
      sel[k] = bi; sv[k] = bv; lg[bi] = -1e30f;
    }
    // softmax over selected logits == renormalized top-k of softmax
    float mx = sv[0], ss = 0.f;
    for (int k = 0; k < H_; ++k){ sv[k] = expf(sv[k] - mx); ss += sv[k]; }
    for (int k = 0; k < H_; ++k){
      topi[(size_t)tok * H_ + k] = sel[k];
      topv[(size_t)tok * H_ + k] = sv[k] / ss;
    }
  }
}

// ---------------------------------------------------------------- LN + routing (FFN, top-2 of 8 -> dense weights)

__global__ __launch_bounds__(256) void ln_route2_kernel(
    const float* __restrict__ x2, const float* __restrict__ g, const float* __restrict__ bb,
    const float* __restrict__ gateF, const int* __restrict__ taskp,
    f16* __restrict__ h2, float* __restrict__ wE)
{
  __shared__ float hrow[D_];
  __shared__ float red[256];
  __shared__ float logits[EF_];
  int tok = blockIdx.x;
  const float* xr = x2 + (size_t)tok * D_;

  float s = 0.f;
  for (int i = threadIdx.x; i < D_; i += 256) s += xr[i];
  float mean = block_sum256(s, red) * (1.0f / D_);
  float vs = 0.f;
  for (int i = threadIdx.x; i < D_; i += 256){ float d = xr[i] - mean; vs += d * d; }
  float var = block_sum256(vs, red) * (1.0f / D_);
  float inv = rsqrtf(var + 1e-5f);
  for (int i = threadIdx.x; i < D_; i += 256){
    float h = (xr[i] - mean) * inv * g[i] + bb[i];
    hrow[i] = h;
    h2[(size_t)tok * D_ + i] = (f16)h;
  }
  __syncthreads();

  int task = taskp[0];
  const float* gF = gateF + (size_t)task * D_ * EF_;
  if (threadIdx.x < EF_){
    float acc = 0.f;
    for (int d = 0; d < D_; ++d) acc += hrow[d] * gF[(size_t)d * EF_ + threadIdx.x];
    logits[threadIdx.x] = acc;
  }
  __syncthreads();
  if (threadIdx.x == 0){
    float lg[EF_];
    for (int i = 0; i < EF_; ++i) lg[i] = logits[i];
    int s0 = 0; float v0 = -1e30f;
    for (int i = 0; i < EF_; ++i) if (lg[i] > v0){ v0 = lg[i]; s0 = i; }
    lg[s0] = -1e30f;
    int s1 = 0; float v1 = -1e30f;
    for (int i = 0; i < EF_; ++i) if (lg[i] > v1){ v1 = lg[i]; s1 = i; }
    float e0 = 1.0f, e1 = expf(v1 - v0), ss = e0 + e1;
    for (int i = 0; i < EF_; ++i) wE[(size_t)tok * EF_ + i] = 0.f;
    wE[(size_t)tok * EF_ + s0] = e0 / ss;
    wE[(size_t)tok * EF_ + s1] = e1 / ss;
  }
}

// ---------------------------------------------------------------- generic tiled WMMA GEMM
// C[M x Nn] = A(f16,row-major) @ B(f32,row-major, per-z slab) [+bias][+residual]
// block = 8 waves covering 32(M) x 64(N); flags: 1 = f16 out, 2 = add residual (f32 out)
// A tile staged via CDNA5 async-to-LDS DMA; B tile staged transposed for vector ds reads.

__global__ __launch_bounds__(256) void gemm_kernel(
    const f16* __restrict__ A, int lda,
    const float* __restrict__ Bw, int ldb, long long strideBz,
    const float* __restrict__ bias,
    void* __restrict__ Cout, int ldc, int coffz,
    const float* __restrict__ resid, int K, int flags)
{
  __shared__ f16 At[32 * 32];        // [M][K]
  __shared__ f16 BtT[64 * 32];       // [N][K] transposed

  int n0 = blockIdx.x * 64;
  int m0 = blockIdx.y * 32;
  int z  = blockIdx.z;
  const float* Bz = Bw + (size_t)z * (size_t)strideBz;

  int w = threadIdx.x >> 5; int mi = w >> 2; int ni = w & 3;
  v8f acc = {};

  for (int kk = 0; kk < K; kk += 32){
    // ---- A tile (f16, no conversion): async DMA global -> LDS, 16B per lane
    if (threadIdx.x < 128){
      int u = threadIdx.x;
      int row = u >> 2, seg = (u & 3) * 8;
      const f16* gsrc = A + (size_t)(m0 + row) * lda + kk + seg;
      unsigned ldst = (unsigned)(size_t)(At + row * 32 + seg);
      asm volatile("global_load_async_to_lds_b128 %0, %1, off"
                   :: "v"(ldst), "v"(gsrc) : "memory");
    }
    // ---- B tile: coalesced f32 row reads, transpose+convert into [N][K] LDS
    {
      int row = threadIdx.x >> 3, c0 = (threadIdx.x & 7) * 8;
      const float* src = Bz + (size_t)(kk + row) * ldb + n0 + c0;
#pragma unroll
      for (int i = 0; i < 8; ++i) BtT[(c0 + i) * 32 + row] = (f16)src[i];
      if (kk + 32 < K) __builtin_prefetch(src + (size_t)32 * ldb, 0, 1);
    }
    asm volatile("s_wait_asynccnt 0x0" ::: "memory");  // issuing waves complete before barrier
    __syncthreads();
    v16h a = frag_a_f16(At + mi * 16 * 32, 32);
    v16h b = frag_b_ldsT(BtT + (ni * 16) * 32);
    acc = wmma16(a, b, acc);
    __syncthreads();
  }

  int lane = threadIdx.x & 31, c = lane & 15, hi = lane >> 4;
  int gcol = z * coffz + n0 + ni * 16 + c;
#pragma unroll
  for (int r = 0; r < 8; ++r){
    int grow = m0 + mi * 16 + r + 8 * hi;
    float val = acc[r];
    if (bias) val += bias[gcol];
    if (flags & 2) val += resid[(size_t)grow * ldc + gcol];
    if (flags & 1) ((f16*)Cout)[(size_t)grow * ldc + gcol] = (f16)val;
    else           ((float*)Cout)[(size_t)grow * ldc + gcol] = val;
  }
}

// ---------------------------------------------------------------- streaming MQA attention
// grid: 768 blocks x 128 thr; wave handles one (batch, head-slot, 16-row i-tile);
// all 4 waves of a block share (batch, head) -> V tile staged transposed in LDS cooperatively.

__global__ __launch_bounds__(128) void attn_kernel(
    const f16* __restrict__ Hall, const f16* __restrict__ kv,
    const int* __restrict__ topi, const float* __restrict__ topv,
    f16* __restrict__ dense)
{
  __shared__ f16 P[4][16 * 32];      // per-wave P transpose buffer
  __shared__ f16 vT[64 * 32];        // [d][j] transposed V tile (shared by 4 waves)
  int wv = threadIdx.x >> 5, lane = threadIdx.x & 31;
  int c = lane & 15, hi = lane >> 4;

  int tile = blockIdx.x * 4 + wv;
  int b   = tile / (H_ * 64);
  int rem = tile % (H_ * 64);
  int hh  = rem / 64;
  int i0  = (rem % 64) * 16;

  // Q fragments (16x64 -> two 16x32 slices), row-dependent expert gather
  int gi = b * N_ + i0 + c;               // c == A-fragment row (M)
  int idxh = topi[(size_t)gi * H_ + hh];
  const f16* qb = Hall + (size_t)gi * HS_ + idxh * HD_;
  v16h a0, a1;
#pragma unroll
  for (int e = 0; e < 16; ++e){
    int kloc = ((e >> 3) << 4) + (hi << 3) + (e & 7);
    a0[e] = qb[kloc];
    a1[e] = qb[32 + kloc];
  }

  float mrow[8], lrow[8];
  v8f zero = {};
  v8f oacc[4];
#pragma unroll
  for (int t = 0; t < 4; ++t) oacc[t] = zero;
#pragma unroll
  for (int r = 0; r < 8; ++r){ mrow[r] = -1e30f; lrow[r] = 0.f; }

  const f16* kbase = kv + (size_t)b * N_ * 128;    // row: [k(64) | v(64)]

  for (int j0 = 0; j0 < N_; j0 += 32){
    // cooperative stage of V tile (32 j-rows x 64 d) transposed -> vT[d][j]
    {
      int t = threadIdx.x;
      int j = t >> 2, seg = (t & 3) * 16;
      const f16* src = kbase + (size_t)(j0 + j) * 128 + 64 + seg;
#pragma unroll
      for (int i = 0; i < 16; ++i) vT[(seg + i) * 32 + j] = src[i];
    }

    v8f slo = zero, shi2 = zero;
    slo  = wmma16(a0, frag_bt_f16(kbase + (size_t)(j0      ) * 128     , 128), slo);
    slo  = wmma16(a1, frag_bt_f16(kbase + (size_t)(j0      ) * 128 + 32, 128), slo);
    shi2 = wmma16(a0, frag_bt_f16(kbase + (size_t)(j0 + 16 ) * 128     , 128), shi2);
    shi2 = wmma16(a1, frag_bt_f16(kbase + (size_t)(j0 + 16 ) * 128 + 32, 128), shi2);

    float corr[8];
#pragma unroll
    for (int r = 0; r < 8; ++r){
      float sl = slo[r] * 0.125f, sh = shi2[r] * 0.125f;
      float v = fmaxf(sl, sh);
      v = fmaxf(v, __shfl_xor(v, 1)); v = fmaxf(v, __shfl_xor(v, 2));
      v = fmaxf(v, __shfl_xor(v, 4)); v = fmaxf(v, __shfl_xor(v, 8));
      float nm = fmaxf(mrow[r], v);
      corr[r] = expf(mrow[r] - nm);
      mrow[r] = nm;
      float pl = expf(sl - nm), ph = expf(sh - nm);
      slo[r] = pl; shi2[r] = ph;
      float rs = pl + ph;
      rs += __shfl_xor(rs, 1); rs += __shfl_xor(rs, 2);
      rs += __shfl_xor(rs, 4); rs += __shfl_xor(rs, 8);
      lrow[r] = lrow[r] * corr[r] + rs;
    }
    // transpose P (C-layout -> A-layout) through per-wave LDS slice
#pragma unroll
    for (int r = 0; r < 8; ++r){
      P[wv][(r + 8 * hi) * 32 + c]      = (f16)slo[r];
      P[wv][(r + 8 * hi) * 32 + 16 + c] = (f16)shi2[r];
    }
    __syncthreads();
    v16h pa;
#pragma unroll
    for (int e = 0; e < 16; ++e){
      int kloc = ((e >> 3) << 4) + (hi << 3) + (e & 7);
      pa[e] = P[wv][(size_t)c * 32 + kloc];
    }
#pragma unroll
    for (int t = 0; t < 4; ++t){
#pragma unroll
      for (int r = 0; r < 8; ++r) oacc[t][r] *= corr[r];
      v16h bv = frag_b_ldsT(vT + (t * 16) * 32);
      oacc[t] = wmma16(pa, bv, oacc[t]);
    }
    __syncthreads();
  }

  // finalize: scale by 1/l and topv, scatter into selected expert slot
#pragma unroll
  for (int r = 0; r < 8; ++r){
    int gtok = b * N_ + i0 + r + 8 * hi;
    int id   = topi[(size_t)gtok * H_ + hh];
    float tv = topv[(size_t)gtok * H_ + hh];
    float sc = tv / lrow[r];
#pragma unroll
    for (int t = 0; t < 4; ++t)
      dense[(size_t)gtok * HS_ + id * HD_ + t * 16 + c] = (f16)(oacc[t][r] * sc);
  }
}

// ---------------------------------------------------------------- fused MoE FFN + shared expert + residual
// one block per 16-token tile; 8 waves; hidden tile (16x1536 f16) lives in LDS;
// zero-weight experts skipped (exact); weight tiles staged transposed per wave.

__global__ __launch_bounds__(256) void ffn_kernel(
    const f16* __restrict__ h2, const float* __restrict__ wE,
    const float* __restrict__ W1f, const float* __restrict__ b1f,
    const float* __restrict__ W2f, const float* __restrict__ b_out,
    const float* __restrict__ Ws1, const float* __restrict__ bs1,
    const float* __restrict__ Ws2, const float* __restrict__ x2,
    float* __restrict__ out)
{
  __shared__ f16 hid[16 * HS_];          // 48 KB
  __shared__ f16 wstage[8][16 * 32];     // 8 KB, per-wave weight transpose buffers
  __shared__ float wr[16][EF_];

  int m0 = blockIdx.x * 16;
  int w = threadIdx.x >> 5, lane = threadIdx.x & 31;
  int c = lane & 15, hi = lane >> 4;
  f16* wb = wstage[w];

  if (threadIdx.x < 128){
    int rr = threadIdx.x >> 3, ee = threadIdx.x & 7;
    wr[rr][ee] = wE[(size_t)(m0 + rr) * EF_ + ee];
  }
  __syncthreads();

  v8f zero = {};
  v8f yac[6];
#pragma unroll
  for (int t = 0; t < 6; ++t) yac[t] = zero;

  const f16* Arow = h2 + (size_t)m0 * D_;

  for (int e = 0; e < EF_ + 1; ++e){
    bool shared_e = (e == EF_);
    const float* W1 = shared_e ? Ws1 : W1f + (size_t)e * D_ * HS_;
    const float* B1 = shared_e ? bs1 : b1f + (size_t)e * HS_;
    const float* W2 = shared_e ? Ws2 : W2f + (size_t)e * HS_ * D_;
    if (!shared_e){
      bool any = false;
      for (int rr = 0; rr < 16; ++rr) if (wr[rr][e] != 0.f) any = true;
      if (!any) continue;               // uniform across block
    }

    // Phase A: hidden = w * gelu(h2_tile @ W1 + b1)  -> LDS (f16)
    for (int t = 0; t < 12; ++t){
      int n1 = (w + t * 8) * 16;
      v8f acc = zero;
      for (int kk = 0; kk < D_; kk += 32){
        v16h a = frag_a_f16(Arow + kk, D_);
        v16h b = frag_b_f32_stage(W1 + (size_t)kk * HS_ + n1, HS_, wb);
        acc = wmma16(a, b, acc);
      }
#pragma unroll
      for (int r = 0; r < 8; ++r){
        int rr = r + 8 * hi, cc = n1 + c;
        float hv = acc[r] + B1[cc];
        float gg = gelu_f(hv);
        float wgt = shared_e ? 1.0f : wr[rr][e];
        hid[(size_t)rr * HS_ + cc] = (f16)(gg * wgt);
      }
    }
    __syncthreads();

    // Phase B: y += hidden @ W2
    for (int t = 0; t < 6; ++t){
      int n2 = (w + t * 8) * 16;
      for (int kk = 0; kk < HS_; kk += 32){
        v16h a = frag_a_f16(hid + kk, HS_);
        v16h b = frag_b_f32_stage(W2 + (size_t)kk * D_ + n2, D_, wb);
        yac[t] = wmma16(a, b, yac[t]);
      }
    }
    __syncthreads();
  }

  // epilogue: out = x2 + y + b_out
#pragma unroll
  for (int t = 0; t < 6; ++t){
    int n2 = (w + t * 8) * 16;
#pragma unroll
    for (int r = 0; r < 8; ++r){
      int gr = m0 + r + 8 * hi, gc = n2 + c;
      out[(size_t)gr * D_ + gc] = x2[(size_t)gr * D_ + gc] + yac[t][r] + b_out[gc];
    }
  }
}

// ---------------------------------------------------------------- host launcher

extern "C" void kernel_launch(void* const* d_in, const int* in_sizes, int n_in,
                              void* d_out, int out_size, void* d_ws, size_t ws_size,
                              hipStream_t stream) {
  (void)in_sizes; (void)n_in; (void)out_size; (void)ws_size;

  const float* x      = (const float*)d_in[0];
  const int*   task   = (const int*)  d_in[1];
  const float* ln1_g  = (const float*)d_in[2];
  const float* ln1_b  = (const float*)d_in[3];
  const float* ln2_g  = (const float*)d_in[4];
  const float* ln2_b  = (const float*)d_in[5];
  const float* gate_a = (const float*)d_in[6];
  const float* Wq_in  = (const float*)d_in[7];
  const float* Wq_out = (const float*)d_in[8];
  const float* W_kv   = (const float*)d_in[9];
  const float* b_kv   = (const float*)d_in[10];
  const float* gate_f = (const float*)d_in[11];
  const float* W1f    = (const float*)d_in[12];
  const float* b1f    = (const float*)d_in[13];
  const float* W2f    = (const float*)d_in[14];
  const float* b_out  = (const float*)d_in[15];
  const float* Ws1    = (const float*)d_in[16];
  const float* bs1    = (const float*)d_in[17];
  const float* Ws2    = (const float*)d_in[18];
  float* out = (float*)d_out;

  // workspace carve (≈50 MB)
  size_t off = 0;
  char* base = (char*)d_ws;
  auto carve = [&](size_t bytes) -> void* {
    void* p = base + off;
    off += (bytes + 255) & ~(size_t)255;
    return p;
  };
  f16*   h1    = (f16*)  carve((size_t)TOK * D_  * 2);
  f16*   Hall  = (f16*)  carve((size_t)TOK * HS_ * 2);   // all-expert Q (EA*HD = 1536)
  f16*   kvbuf = (f16*)  carve((size_t)TOK * 128 * 2);
  int*   topi  = (int*)  carve((size_t)TOK * H_  * 4);
  float* topv  = (float*)carve((size_t)TOK * H_  * 4);
  f16*   dense = (f16*)  carve((size_t)TOK * HS_ * 2);
  float* x2    = (float*)carve((size_t)TOK * D_  * 4);
  f16*   h2    = (f16*)  carve((size_t)TOK * D_  * 2);
  float* wE    = (float*)carve((size_t)TOK * EF_ * 4);

  // 1. LN1 + attention routing
  ln_route1_kernel<<<TOK, 256, 0, stream>>>(x, ln1_g, ln1_b, gate_a, task, h1, topi, topv);

  // 2. all-expert Q projections: Hall[:, e*64:(e+1)*64] = h1 @ Wq_in[e]
  gemm_kernel<<<dim3(1, TOK/32, EA_), 256, 0, stream>>>(
      h1, D_, Wq_in, HD_, (long long)D_ * HD_, nullptr, Hall, HS_, HD_, nullptr, D_, 1);

  // 3. KV projection (+bias)
  gemm_kernel<<<dim3(2, TOK/32, 1), 256, 0, stream>>>(
      h1, D_, W_kv, 128, 0, b_kv, kvbuf, 128, 0, nullptr, D_, 1);

  // 4. zero the expert-scatter buffer, then streaming attention scatters into it
  hipMemsetAsync(dense, 0, (size_t)TOK * HS_ * 2, stream);
  attn_kernel<<<(B_*H_*64)/4, 128, 0, stream>>>(Hall, kvbuf, topi, topv, dense);

  // 5. output projection + residual: x2 = x + dense @ Wq_out_flat
  gemm_kernel<<<dim3(D_/64, TOK/32, 1), 256, 0, stream>>>(
      dense, HS_, Wq_out, D_, 0, nullptr, x2, D_, 0, x, HS_, 2);

  // 6. LN2 + FFN routing
  ln_route2_kernel<<<TOK, 256, 0, stream>>>(x2, ln2_g, ln2_b, gate_f, task, h2, wE);

  // 7. fused MoE FFN + shared expert + b_out + residual -> out
  ffn_kernel<<<TOK/16, 256, 0, stream>>>(h2, wE, W1f, b1f, W2f, b_out, Ws1, bs1, Ws2, x2, out);
}
